// GAMENet_63934883169128
// MI455X (gfx1250) — matrix operand order repaired and checked by reading, not compiled, not run on hardware
//
#include <hip/hip_runtime.h>

// ---------------- CDNA5 WMMA types ----------------
typedef __attribute__((ext_vector_type(16))) __bf16 v16bf;
typedef __attribute__((ext_vector_type(8)))  float  v8f;
typedef __attribute__((ext_vector_type(4)))  unsigned int u32x4;

struct FragB { u32x4 lo, hi; };   // 32 bytes = 16 bf16 = one WMMA A/B operand

static __device__ inline v16bf frag_cast(FragB f) {
    return __builtin_bit_cast(v16bf, f);
}

static __device__ inline v8f wmma_bf16(v16bf a, v16bf b, v8f c) {
    // v_wmma_f32_16x16x32_bf16: D = A(16x32) x B(32x16) + C
    return __builtin_amdgcn_wmma_f32_16x16x32_bf16(false, a, false, b, (short)0, c, false, false);
}

// ---------------- Generic bf16 WMMA GEMM ----------------
// C[M,N] = act(A[M,K] @ B + bias).  A: M x K row-major bf16.  B: stored N x K ("transposed") bf16
// so each lane's fragment is 32 contiguous bytes.  Output: f32 (Cf) and/or bf16 (Cb, optionally
// stored transposed for chaining as a future B operand).  One wave per 16x16 tile.
__global__ __launch_bounds__(32) void gemm_bf16_wmma(
    const __bf16* __restrict__ A, int lda,
    const __bf16* __restrict__ Bn, int ldb,
    const float* __restrict__ bias,
    float* __restrict__ Cf, __bf16* __restrict__ Cb, int transCb,
    int ldc, int K, int relu)
{
    const int lane = threadIdx.x;
    const int l16 = lane & 15, half = lane >> 4;
    const int rowA = blockIdx.y * 16 + l16;
    const int colB = blockIdx.x * 16 + l16;

    v8f acc;
#pragma unroll
    for (int i = 0; i < 8; ++i) acc[i] = 0.f;

    const u32x4* pa = (const u32x4*)(A + (size_t)rowA * lda);   // 8 bf16 per u32x4
    const u32x4* pb = (const u32x4*)(Bn + (size_t)colB * ldb);

    const int K8 = K >> 3;
    for (int k8 = 0; k8 < K8; k8 += 4) {       // 32 elements of K per iteration
        FragB fa, fb;
        fa.lo = pa[k8 + half * 2];             // k = k0 + half*16 .. +7
        fa.hi = pa[k8 + half * 2 + 1];         // k = k0 + half*16 + 8 .. +15
        fb.lo = pb[k8 + half * 2];             // k = k0 + half*16 .. +7
        fb.hi = pb[k8 + half * 2 + 1];         // k = k0 + half*16 + 8 .. +15
        __builtin_prefetch(pa + k8 + 4, 0, 1);
        __builtin_prefetch(pb + k8 + 4, 0, 1);
        acc = wmma_bf16(frag_cast(fa), frag_cast(fb), acc);
    }

    const float bv = bias ? bias[colB] : 0.f;
#pragma unroll
    for (int r = 0; r < 8; ++r) {
        int row = blockIdx.y * 16 + r + 8 * half;
        float v = acc[r] + bv;
        if (relu) v = v > 0.f ? v : 0.f;
        if (Cf) Cf[(size_t)row * ldc + colB] = v;
        if (Cb) {
            if (transCb) Cb[(size_t)colB * ldc + row] = (__bf16)v;
            else         Cb[(size_t)row * ldc + colB] = (__bf16)v;
        }
    }
}

// ---------------- Fused GRU step ----------------
// h, day_rep, o1, xp all bf16.  Wh is pre-packed into WMMA B-fragment order:
// WhP[((jc*3+gate)*4+ks)*32 + lane] is one 32-byte fragment -> inner loop is pure b128 loads + wmma.
// 8 chunks x 3 gates x 4 ksteps = 96 v_wmma per wave.  x-projection fused (GRU1: 2 FMAs from
// glu/time; GRU2: bf16 reads of precomputed xp).  day_rep scattered at t == length-1.
__global__ __launch_bounds__(32) void gru_step_wmma(
    const __bf16* __restrict__ hold, __bf16* __restrict__ hnew,
    const FragB* __restrict__ WhP, const float* __restrict__ bh,
    const float* __restrict__ glu, const float* __restrict__ timev,
    const float* __restrict__ Wx, const float* __restrict__ bx,
    const __bf16* __restrict__ xp, int xprs, int xpss,
    int t, int T,
    const int* __restrict__ length, __bf16* __restrict__ day_rep,
    __bf16* __restrict__ seq_out, int seq_stride)
{
    const int lane = threadIdx.x;
    const int l16 = lane & 15, half = lane >> 4;
    const int rowbase = blockIdx.x * 16;
    const int E = 128, E3 = 384;

    v16bf afr[4];
    {
        const u32x4* pa = (const u32x4*)(hold + (size_t)(rowbase + l16) * E);
#pragma unroll
        for (int ks = 0; ks < 4; ++ks) {
            FragB f;
            f.lo = pa[ks * 4 + half * 2];
            f.hi = pa[ks * 4 + half * 2 + 1];
            afr[ks] = frag_cast(f);
        }
    }

#pragma unroll
    for (int jc = 0; jc < 8; ++jc) {
        v8f aR, aZ, aN;
#pragma unroll
        for (int i = 0; i < 8; ++i) { aR[i] = 0.f; aZ[i] = 0.f; aN[i] = 0.f; }

#pragma unroll
        for (int ks = 0; ks < 4; ++ks) {
            v16bf br = frag_cast(WhP[((jc * 3 + 0) * 4 + ks) * 32 + lane]);
            v16bf bz = frag_cast(WhP[((jc * 3 + 1) * 4 + ks) * 32 + lane]);
            v16bf bn = frag_cast(WhP[((jc * 3 + 2) * 4 + ks) * 32 + lane]);
            aR = wmma_bf16(afr[ks], br, aR);
            aZ = wmma_bf16(afr[ks], bz, aZ);
            aN = wmma_bf16(afr[ks], bn, aN);
        }

        const int col = jc * 16 + l16;
        const float bhr = bh[col], bhz = bh[col + 128], bhn = bh[col + 256];
        float w0r = 0, w1r = 0, w0z = 0, w1z = 0, w0n = 0, w1n = 0, bxr = 0, bxz = 0, bxn = 0;
        if (glu) {
            w0r = Wx[col];       w1r = Wx[E3 + col];
            w0z = Wx[col + 128]; w1z = Wx[E3 + col + 128];
            w0n = Wx[col + 256]; w1n = Wx[E3 + col + 256];
            bxr = bx[col]; bxz = bx[col + 128]; bxn = bx[col + 256];
        }
#pragma unroll
        for (int r = 0; r < 8; ++r) {
            int row = rowbase + r + 8 * half;
            float xr, xz, xn;
            if (glu) {
                float g = glu[row * T + t], tv = timev[row * T + t];
                xr = g * w0r + tv * w1r + bxr;
                xz = g * w0z + tv * w1z + bxz;
                xn = g * w0n + tv * w1n + bxn;
            } else {
                const __bf16* xrow = xp + (size_t)row * xprs + t * xpss;
                xr = (float)xrow[col]; xz = (float)xrow[col + 128]; xn = (float)xrow[col + 256];
            }
            float rg = 1.f / (1.f + __expf(-(xr + aR[r] + bhr)));
            float zg = 1.f / (1.f + __expf(-(xz + aZ[r] + bhz)));
            float ng = tanhf(xn + rg * (aN[r] + bhn));
            float hp = (float)hold[(size_t)row * E + col];
            float hv = (1.f - zg) * ng + zg * hp;
            hnew[(size_t)row * E + col] = (__bf16)hv;
            if (length && (length[row] - 1) == t) day_rep[(size_t)row * E + col] = (__bf16)hv;
            if (seq_out) seq_out[(size_t)row * seq_stride + t * E + col] = (__bf16)hv;
        }
    }
}

// ---------------- converters / packers ----------------
__global__ void cvt_bf16(const float* __restrict__ in, __bf16* __restrict__ out, int n) {
    int i = blockIdx.x * blockDim.x + threadIdx.x;
    if (i < n) out[i] = (__bf16)in[i];
}

// in: R x C f32 -> out: C x R bf16 (B operands stored N x K)
__global__ void cvt_bf16_T(const float* __restrict__ in, __bf16* __restrict__ out, int R, int C) {
    int idx = blockIdx.x * blockDim.x + threadIdx.x;
    if (idx >= R * C) return;
    int r = idx / C, c = idx % C;
    out[c * R + r] = (__bf16)in[idx];
}

// Pack Wh (128 x 384 f32) into WMMA B-fragment order (see gru_step_wmma).
__global__ void pack_wh(const float* __restrict__ Wh, __bf16* __restrict__ out) {
    int idx = blockIdx.x * blockDim.x + threadIdx.x;   // 128*384 = 49152
    if (idx >= 128 * 384) return;
    int e = idx & 15;
    int lane = (idx >> 4) & 31;
    int rest = idx >> 9;
    int ks = rest & 3; rest >>= 2;
    int g = rest % 3;
    int jc = rest / 3;
    int k = ks * 32 + (lane >> 4) * 16 + e;            // fragment element e <-> k = half*16 + e
    int col = g * 128 + jc * 16 + (lane & 15);
    out[idx] = (__bf16)Wh[k * 384 + col];
}

__global__ void fill_bf16_zero(__bf16* p, int n) {
    int i = blockIdx.x * blockDim.x + threadIdx.x;
    if (i < n) p[i] = (__bf16)0.f;
}

// ---------------- small elementwise kernels ----------------
__global__ void dm_combine(const float* __restrict__ dmE, const float* __restrict__ dmD,
                           const float* __restrict__ inter,
                           __bf16* __restrict__ dm_nk, __bf16* __restrict__ dmT) {
    int i = blockIdx.x * blockDim.x + threadIdx.x;     // 512*128
    if (i >= 512 * 128) return;
    float v = dmE[i] - inter[0] * dmD[i];
    int m = i >> 7, c = i & 127;
    dm_nk[i] = (__bf16)v;          // natural 512x128  (N x K for the scores GEMM)
    dmT[c * 512 + m] = (__bf16)v;  // transposed 128x512 (N x K for fact GEMMs)
}

__global__ void copy_query_to_hcat(const __bf16* __restrict__ q, __bf16* __restrict__ hcat) {
    int i = blockIdx.x * blockDim.x + threadIdx.x;     // 512*128
    int b = i >> 7, c = i & 127;
    hcat[(size_t)b * 512 + 128 + c] = q[i];
}

// scores[b,v] = dot(query[b,:], o1[b,v,:]) for v in [0, Vm1)
__global__ void vscore_kernel(const __bf16* __restrict__ query, const __bf16* __restrict__ o1,
                              float* __restrict__ vw, int Vm1) {
    int idx = blockIdx.x * blockDim.x + threadIdx.x;
    if (idx >= 512 * Vm1) return;
    int b = idx / Vm1, v = idx % Vm1;
    const __bf16* q = query + (size_t)b * 128;
    const __bf16* o = o1 + ((size_t)b * 24 + v) * 128;
    float s = 0.f;
#pragma unroll 8
    for (int k = 0; k < 128; ++k) s += (float)q[k] * (float)o[k];
    vw[idx] = s;
}

__global__ void vsoftmax_kernel(float* vw, int Vm1) {
    int b = blockIdx.x;
    int lane = threadIdx.x;   // 32
    float v = (lane < Vm1) ? vw[b * Vm1 + lane] : -3.4e38f;
    float m = v;
    for (int off = 16; off; off >>= 1) m = fmaxf(m, __shfl_xor(m, off, 32));
    float e = (lane < Vm1) ? __expf(v - m) : 0.f;
    float s = e;
    for (int off = 16; off; off >>= 1) s += __shfl_xor(s, off, 32);
    if (lane < Vm1) vw[b * Vm1 + lane] = e / s;
}

// weighted[b,m] = sum_v vw[b,v] * med[b,v,m]   (bf16 out -> A operand of fact2 GEMM)
__global__ void weighted_kernel(const float* __restrict__ vw, const float* __restrict__ med,
                                __bf16* __restrict__ out, int Vm1) {
    int idx = blockIdx.x * blockDim.x + threadIdx.x;
    if (idx >= 512 * 512) return;
    int b = idx >> 9, m = idx & 511;
    float s = 0.f;
    for (int v = 0; v < Vm1; ++v) s += vw[b * Vm1 + v] * med[((size_t)b * 24 + v) * 512 + m];
    out[idx] = (__bf16)s;
}

// row softmax over 512 f32 columns -> bf16 output (A operand of fact1 GEMM)
__global__ void row_softmax512(const float* __restrict__ x, __bf16* __restrict__ out) {
    const float* p = x + (size_t)blockIdx.x * 512;
    __bf16* po = out + (size_t)blockIdx.x * 512;
    int tid = threadIdx.x;  // 256
    __shared__ float red[8];
    __shared__ float red2[8];
    float v0 = p[tid], v1 = p[tid + 256];
    float m = fmaxf(v0, v1);
    for (int off = 16; off; off >>= 1) m = fmaxf(m, __shfl_xor(m, off, 32));
    if ((tid & 31) == 0) red[tid >> 5] = m;
    __syncthreads();
    float mm = red[0];
#pragma unroll
    for (int i = 1; i < 8; ++i) mm = fmaxf(mm, red[i]);
    float e0 = __expf(v0 - mm), e1 = __expf(v1 - mm);
    float s = e0 + e1;
    for (int off = 16; off; off >>= 1) s += __shfl_xor(s, off, 32);
    if ((tid & 31) == 0) red2[tid >> 5] = s;
    __syncthreads();
    float ss = 0.f;
#pragma unroll
    for (int i = 0; i < 8; ++i) ss += red2[i];
    float inv = 1.f / ss;
    po[tid] = (__bf16)(e0 * inv);
    po[tid + 256] = (__bf16)(e1 * inv);
}

// ---------------- Host orchestration ----------------
extern "C" void kernel_launch(void* const* d_in, const int* in_sizes, int n_in,
                              void* d_out, int out_size, void* d_ws, size_t ws_size,
                              hipStream_t stream) {
    (void)in_sizes; (void)n_in; (void)out_size; (void)ws_size;
    const int B = 512, V = 24, T = 25, E = 128, MED = 512, LAB = 512;
    const int BV = B * V;         // 12288
    const int E3 = 3 * E;         // 384

    const float* lab      = (const float*)d_in[0];
    const float* glu      = (const float*)d_in[1];
    const float* timev    = (const float*)d_in[2];
    const float* med      = (const float*)d_in[3];
    const int*   length   = (const int*)d_in[4];
    const float* adj_ehr  = (const float*)d_in[5];
    const float* adj_ddi  = (const float*)d_in[6];
    const float* lab_embw = (const float*)d_in[7];
    const float* Ws1      = (const float*)d_in[8];
    const float* bs1      = (const float*)d_in[9];
    const float* Ws2      = (const float*)d_in[10];
    const float* bs2      = (const float*)d_in[11];
    const float* Wxg      = (const float*)d_in[12];
    const float* Whg      = (const float*)d_in[13];
    const float* bxg      = (const float*)d_in[14];
    const float* bhg      = (const float*)d_in[15];
    const float* Wxe      = (const float*)d_in[16];
    const float* Whe      = (const float*)d_in[17];
    const float* bxe      = (const float*)d_in[18];
    const float* bhe      = (const float*)d_in[19];
    const float* Wq       = (const float*)d_in[20];
    const float* bq       = (const float*)d_in[21];
    const float* We1      = (const float*)d_in[22];
    const float* be1      = (const float*)d_in[23];
    const float* We2      = (const float*)d_in[24];
    const float* be2      = (const float*)d_in[25];
    const float* Wd1      = (const float*)d_in[26];
    const float* bd1      = (const float*)d_in[27];
    const float* Wd2      = (const float*)d_in[28];
    const float* bd2      = (const float*)d_in[29];
    const float* inter    = (const float*)d_in[30];
    const float* Wo1      = (const float*)d_in[31];
    const float* bo1      = (const float*)d_in[32];
    const float* Wo2      = (const float*)d_in[33];
    const float* bo2      = (const float*)d_in[34];
    float* out = (float*)d_out;

    // byte-based workspace carve-up, 256B aligned
    char* base = (char*)d_ws;
    size_t off = 0;
    auto allocBytes = [&](size_t bytes) {
        off = (off + 255) & ~(size_t)255;
        void* p = base + off;
        off += bytes;
        return p;
    };
    auto allocBF = [&](size_t n) { return (__bf16*)allocBytes(n * 2); };
    auto allocF  = [&](size_t n) { return (float*)allocBytes(n * 4); };

    __bf16* hA    = allocBF((size_t)BV * E);
    __bf16* hB    = allocBF((size_t)BV * E);
    __bf16* day   = allocBF((size_t)BV * E);
    __bf16* xp2   = allocBF((size_t)BV * E3);
    __bf16* o1    = allocBF((size_t)B * V * E);
    __bf16* labB  = allocBF((size_t)B * LAB);
    __bf16* adjEb = allocBF((size_t)MED * MED);
    __bf16* adjDb = allocBF((size_t)MED * MED);
    __bf16* lembT = allocBF((size_t)LAB * E);
    __bf16* Ws1T  = allocBF((size_t)E * 2 * E);
    __bf16* Ws2T  = allocBF((size_t)2 * E * E);
    __bf16* WxeT  = allocBF((size_t)E * E3);
    __bf16* WqT   = allocBF((size_t)E * E);
    __bf16* We1T  = allocBF((size_t)MED * E);
    __bf16* We2T  = allocBF((size_t)E * E);
    __bf16* Wd1T  = allocBF((size_t)MED * E);
    __bf16* Wd2T  = allocBF((size_t)E * E);
    __bf16* Wo1T  = allocBF((size_t)4 * E * E3);
    __bf16* Wo2T  = allocBF((size_t)E3 * MED);
    __bf16* WhgP  = allocBF((size_t)E * E3);
    __bf16* WheP  = allocBF((size_t)E * E3);
    __bf16* lab_e = allocBF((size_t)B * E);
    __bf16* s1    = allocBF((size_t)B * 2 * E);
    __bf16* hcat  = allocBF((size_t)B * 4 * E);
    __bf16* gA    = allocBF((size_t)MED * E);
    __bf16* gTt   = allocBF((size_t)E * MED);
    float*  dmE   = allocF((size_t)MED * E);
    float*  dmD   = allocF((size_t)MED * E);
    __bf16* dm_nk = allocBF((size_t)MED * E);
    __bf16* dmT   = allocBF((size_t)E * MED);
    __bf16* query = allocBF((size_t)B * E);
    float*  scores= allocF((size_t)B * MED);
    __bf16* kw1   = allocBF((size_t)B * MED);
    float*  vw    = allocF((size_t)B * (V - 1));
    __bf16* wtd   = allocBF((size_t)B * MED);
    __bf16* ho    = allocBF((size_t)B * E3);

    auto gemm = [&](const __bf16* A, int lda, const __bf16* Bn, int ldb,
                    const float* bias, float* Cf, __bf16* Cb, int transCb,
                    int ldc, int M, int N, int K, int relu) {
        dim3 g(N / 16, M / 16);
        gemm_bf16_wmma<<<g, dim3(32), 0, stream>>>(A, lda, Bn, ldb, bias, Cf, Cb, transCb, ldc, K, relu);
    };
    auto cvt = [&](const float* in, __bf16* o, int n) {
        cvt_bf16<<<(n + 255) / 256, 256, 0, stream>>>(in, o, n);
    };
    auto cvtT = [&](const float* in, __bf16* o, int R, int C) {
        cvt_bf16_T<<<(R * C + 255) / 256, 256, 0, stream>>>(in, o, R, C);
    };

    // ---- one-shot conversions / packs ----
    cvt(lab, labB, B * LAB);
    cvt(adj_ehr, adjEb, MED * MED);
    cvt(adj_ddi, adjDb, MED * MED);
    cvtT(lab_embw, lembT, LAB, E);
    cvtT(Ws1, Ws1T, E, 2 * E);
    cvtT(Ws2, Ws2T, 2 * E, E);
    cvtT(Wxe, WxeT, E, E3);
    cvtT(Wq, WqT, E, E);
    cvtT(We1, We1T, MED, E);
    cvtT(We2, We2T, E, E);
    cvtT(Wd1, Wd1T, MED, E);
    cvtT(Wd2, Wd2T, E, E);
    cvtT(Wo1, Wo1T, 4 * E, E3);
    cvtT(Wo2, Wo2T, E3, MED);
    pack_wh<<<(E * E3 + 255) / 256, 256, 0, stream>>>(Whg, WhgP);
    pack_wh<<<(E * E3 + 255) / 256, 256, 0, stream>>>(Whe, WheP);

    // ---- GRU1: 12288 sequences, T=25, fully fused step ----
    fill_bf16_zero<<<(BV * E + 255) / 256, 256, 0, stream>>>(hA, BV * E);
    for (int t = 0; t < T; ++t) {
        __bf16* src = (t & 1) ? hB : hA;
        __bf16* dst = (t & 1) ? hA : hB;
        gru_step_wmma<<<BV / 16, 32, 0, stream>>>(
            src, dst, (const FragB*)WhgP, bhg,
            glu, timev, Wxg, bxg,
            nullptr, 0, 0, t, T,
            length, day, nullptr, 0);
    }

    // ---- GRU2: x-proj as one GEMM, then 24 fused steps recording o1 ----
    gemm(day, E, WxeT, E, bxe, nullptr, xp2, 0, E3, BV, E3, E, 0);
    fill_bf16_zero<<<(B * E + 255) / 256, 256, 0, stream>>>(hA, B * E);
    for (int t = 0; t < V; ++t) {
        __bf16* src = (t & 1) ? hB : hA;
        __bf16* dst = (t & 1) ? hA : hB;
        gru_step_wmma<<<B / 16, 32, 0, stream>>>(
            src, dst, (const FragB*)WheP, bhe,
            nullptr, nullptr, nullptr, nullptr,
            xp2, V * E3, E3, t, V,
            nullptr, nullptr, o1, V * E);
    }

    // ---- drug memory: two GCNs ----
    gemm(adjEb, MED, We1T, MED, be1, nullptr, gA, 0, E, MED, E, MED, 1);
    gemm(gA, E, We2T, E, nullptr, nullptr, gTt, 1, MED, MED, E, E, 0);   // store transposed (E x MED)
    gemm(adjEb, MED, gTt, MED, be2, dmE, nullptr, 0, E, MED, E, MED, 0);
    gemm(adjDb, MED, Wd1T, MED, bd1, nullptr, gA, 0, E, MED, E, MED, 1);
    gemm(gA, E, Wd2T, E, nullptr, nullptr, gTt, 1, MED, MED, E, E, 0);
    gemm(adjDb, MED, gTt, MED, bd2, dmD, nullptr, 0, E, MED, E, MED, 0);
    dm_combine<<<(MED * E + 255) / 256, 256, 0, stream>>>(dmE, dmD, inter, dm_nk, dmT);

    // ---- static path -> hcat[:, 0:128] ----
    gemm(labB, LAB, lembT, LAB, nullptr, nullptr, lab_e, 0, E, B, E, LAB, 0);
    gemm(lab_e, E, Ws1T, E, bs1, nullptr, s1, 0, 2 * E, B, 2 * E, E, 1);
    gemm(s1, 2 * E, Ws2T, 2 * E, bs2, nullptr, hcat, 0, 4 * E, B, E, 2 * E, 1);

    // ---- query = relu(o1[:, -1] @ Wq + bq) -> hcat[:, 128:256] ----
    gemm(o1 + (size_t)(V - 1) * E, V * E, WqT, E, bq, nullptr, query, 0, E, B, E, E, 1);
    copy_query_to_hcat<<<(B * E + 255) / 256, 256, 0, stream>>>(query, hcat);

    // ---- fact1 = softmax(query @ dm^T) @ dm -> hcat[:, 256:384] ----
    gemm(query, E, dm_nk, E, nullptr, scores, nullptr, 0, MED, B, MED, E, 0);
    row_softmax512<<<B, 256, 0, stream>>>(scores, kw1);
    gemm(kw1, MED, dmT, MED, nullptr, nullptr, hcat + 2 * E, 0, 4 * E, B, E, MED, 1);

    // ---- fact2 -> hcat[:, 384:512] ----
    vscore_kernel<<<(B * (V - 1) + 255) / 256, 256, 0, stream>>>(query, o1, vw, V - 1);
    vsoftmax_kernel<<<B, 32, 0, stream>>>(vw, V - 1);
    weighted_kernel<<<(B * MED + 255) / 256, 256, 0, stream>>>(vw, med, wtd, V - 1);
    gemm(wtd, MED, dmT, MED, nullptr, nullptr, hcat + 3 * E, 0, 4 * E, B, E, MED, 1);

    // ---- output MLP ----
    gemm(hcat, 4 * E, Wo1T, 4 * E, bo1, nullptr, ho, 0, E3, B, E3, 4 * E, 1);
    gemm(ho, E3, Wo2T, E3, bo2, out, nullptr, 0, MED, B, MED, E3, 0);
}
